// proj_net_90340342104672
// MI455X (gfx1250) — matrix operand order repaired and checked by reading, not compiled, not run on hardware
//
#include <hip/hip_runtime.h>
#include <hip/hip_bf16.h>

// ---------------------------------------------------------------------------
// Fused RNN for MI455X (gfx1250, wave32, WMMA bf16 16x16x32, f32 accum)
//
//   B=256, T=2048, I=10, H=512, O=9
//   h_{t} = relu(x_t @ W_in^T + h_{t-1} @ W_rec^T);  y_t = h_t @ W_out^T + b
//
// 16 workgroups (one per 16-batch-row slice), 256 threads = 8 waves each.
// Wave w owns N-tiles {4w..4w+3} of H. Per step per wave:
//   4 WMMA (input proj, K padded 10->32) + 64 WMMA (recurrent, K=512)
//   + 2 WMMA (output proj partial on its own h_new columns)  => 70 wmma/wave/step
// W_rec^T fragments: K rows 0..255 resident in LDS (256KB), 256..511 from L2.
// ---------------------------------------------------------------------------

typedef __attribute__((ext_vector_type(16))) __bf16       v16bf;
typedef __attribute__((ext_vector_type(8)))  float        v8f;
typedef __attribute__((ext_vector_type(4)))  unsigned int u32x4;

#define B_   256
#define T_   2048
#define I_   10
#define H_   512
#define O_   9

// workspace layout (bytes): pre-swizzled bf16 B-fragments, 1KB per 32x16 tile
#define WS_WREC 0          // 16 kit * 32 nt * 1024 = 524288
#define WS_WIN  524288     //  1 kit * 32 nt * 1024 =  32768
#define WS_WOUT 557056     // 16 kit *  1 nt * 1024 =  16384

// LDS layout (bytes)
#define LDS_BRES 0         // W_rec fragments, kits 0..7: 262144
#define LDS_H    262144    // 2 * (16 rows * 1040 B)    :  33280 (bank-friendly stride)
#define LDS_XIN  295424    // 2 * 16*16*f32             :   2048
#define LDS_YP   297472    // 2 * 8 waves*16*16*f32     :  16384
#define LDS_TOTAL 313856   // 306.5 KB of 320 KB

#define HROW 1040          // 520 bf16: 512 cols + 8 pad -> conflict-free ds_b128

static __device__ __forceinline__ unsigned short f32_to_bf16_bits(float f) {
  union { __bf16 b; unsigned short s; } u;
  u.b = (__bf16)f;          // RNE convert
  return u.s;
}

// ---------------------------------------------------------------------------
// prep: scatter W_rec^T / W_in^T / W_out^T into CDNA5 bf16 B-fragment layout.
// B 32x16 tile (contiguous k as used here, not ISA table order): lane<16 -> N=lane,  K=0..15 ; lane>=16 -> K=16..31.
// Stored as two 512B chunks per tile: [elems 0..7 of all lanes][elems 8..15].
// ---------------------------------------------------------------------------
__global__ void prep_frags(const float* __restrict__ W_in,
                           const float* __restrict__ W_rec,
                           const float* __restrict__ W_out,
                           unsigned char* __restrict__ ws) {
  const int gid  = blockIdx.x * 256 + threadIdx.x;   // one (tile,lane) per thread
  const int lane = gid & 31;
  const int n16  = lane & 15;
  const int kb16 = (lane >> 4) << 4;                 // 0 or 16
  unsigned short e[16];
  unsigned char* dst;

  if (gid < 16384) {                                 // W_rec^T: B[k][n] = W_rec[n][k]
    const int nt = (gid >> 5) & 31, kit = gid >> 10;
    const int n = nt * 16 + n16, kbase = kit * 32 + kb16;
#pragma unroll
    for (int i = 0; i < 16; ++i) e[i] = f32_to_bf16_bits(W_rec[n * H_ + kbase + i]);
    dst = ws + WS_WREC + (kit * 32 + nt) * 1024;
  } else if (gid < 17408) {                          // W_in^T, K padded 10->32
    const int nt = (gid - 16384) >> 5;
    const int n = nt * 16 + n16;
#pragma unroll
    for (int i = 0; i < 16; ++i) {
      const int k = kb16 + i;
      e[i] = (k < I_) ? f32_to_bf16_bits(W_in[n * I_ + k]) : (unsigned short)0;
    }
    dst = ws + WS_WIN + nt * 1024;
  } else if (gid < 17920) {                          // W_out^T, N padded 9->16
    const int kit = (gid - 17408) >> 5;
    const int kbase = kit * 32 + kb16;
#pragma unroll
    for (int i = 0; i < 16; ++i)
      e[i] = (n16 < O_) ? f32_to_bf16_bits(W_out[n16 * H_ + kbase + i]) : (unsigned short)0;
    dst = ws + WS_WOUT + kit * 1024;
  } else {
    return;
  }
  union { unsigned short s[8]; u32x4 u; } c0, c1;
#pragma unroll
  for (int i = 0; i < 8; ++i) { c0.s[i] = e[i]; c1.s[i] = e[8 + i]; }
  *(u32x4*)(dst + lane * 16)       = c0.u;
  *(u32x4*)(dst + 512 + lane * 16) = c1.u;
}

// ---------------------------------------------------------------------------
union BFrag { u32x4 u[2]; v16bf v; };

// B fragment: two b128 loads covering a contiguous 1KB tile (global or LDS)
static __device__ __forceinline__ v16bf load_bfrag(const unsigned char* p, int lane) {
  BFrag f;
  f.u[0] = *(const u32x4*)(p + lane * 16);
  f.u[1] = *(const u32x4*)(p + 512 + lane * 16);
  return f.v;
}

// A fragment 16x32 from the row-major bf16 h buffer (contiguous k matching B, not ISA table order):
// lane<16: M=lane, K=c..c+15; lane>=16: M=lane-16, K=c+16..c+31.
static __device__ __forceinline__ v16bf load_afrag(const unsigned char* hbase,
                                                   int mrow, int hi, int kit) {
  BFrag f;
  const unsigned char* p = hbase + mrow * HROW + kit * 64 + hi * 32;
  f.u[0] = *(const u32x4*)(p);
  f.u[1] = *(const u32x4*)(p + 16);
  return f.v;
}

__global__ __launch_bounds__(256, 1)
void rnn_fused(const float* __restrict__ inputs,
               const float* __restrict__ b_out,
               const unsigned char* __restrict__ ws,
               float* __restrict__ out) {
  extern __shared__ unsigned char sm[];
  const int tid  = threadIdx.x;
  const int lane = tid & 31;
  const int w    = tid >> 5;        // wave 0..7
  const int m0   = blockIdx.x * 16; // batch-row slice
  const int mrow = lane & 15;
  const int hi   = lane >> 4;       // half-wave select
  const int ntb  = w * 4;           // first owned N tile

  // one-time: resident W_rec fragments (K=0..255) into LDS
  {
    const u32x4* src = (const u32x4*)(ws + WS_WREC);
    u32x4*       dst = (u32x4*)(sm + LDS_BRES);
    for (int i = tid; i < 262144 / 16; i += 256) dst[i] = src[i];
  }
  {  // h0 = 0 ; zero both xin buffers (pad cols stay 0 forever)
    unsigned int* h0 = (unsigned int*)(sm + LDS_H);
    for (int i = tid; i < 16 * HROW / 4; i += 256) h0[i] = 0u;
    unsigned int* xz = (unsigned int*)(sm + LDS_XIN);
    for (int i = tid; i < 2048 / 4; i += 256) xz[i] = 0u;
  }
  if (tid < 160) {  // stage input rows for t=0
    const int r = tid / 10, i = tid % 10;
    *(float*)(sm + LDS_XIN + (r * 16 + i) * 4) = inputs[(size_t)(m0 + r) * T_ * I_ + i];
  }
  const float bias = (tid < 144) ? b_out[tid % O_] : 0.f;
  __syncthreads();

  for (int t = 0; t < T_; ++t) {
    const int cur = t & 1, nxt = cur ^ 1;

    // ---- finalize y for step t-1 (partials live in ypart[nxt]) ------------
    if (t > 0 && tid < 144) {
      const int m = tid / O_, o = tid % O_;
      const float* yp = (const float*)(sm + LDS_YP + nxt * 8192);
      float s = bias;
#pragma unroll
      for (int ww = 0; ww < 8; ++ww) s += yp[ww * 256 + m * 16 + o];
      out[((size_t)(m0 + m) * T_ + (t - 1)) * O_ + o] = s;
    }

    // ---- A fragment of the input tile (K padded 10->32) -------------------
    v16bf axin;
#pragma unroll
    for (int i = 0; i < 16; ++i) axin[i] = (__bf16)0.f;
    {
      const float* xr = (const float*)(sm + LDS_XIN + cur * 1024) + mrow * 16;
      if (hi == 0) {
#pragma unroll
        for (int i = 0; i < 10; ++i) axin[i] = (__bf16)xr[i];  // K=0..9
      }
    }

    // ---- acc = x_t @ W_in^T  (accumulator init via WMMA, C=0) -------------
    v8f acc[4];
#pragma unroll
    for (int j = 0; j < 4; ++j) {
      v16bf b = load_bfrag(ws + WS_WIN + (ntb + j) * 1024, lane);
      v8f z = {0.f, 0.f, 0.f, 0.f, 0.f, 0.f, 0.f, 0.f};
      acc[j] = __builtin_amdgcn_wmma_f32_16x16x32_bf16(false, axin, false, b,
                                                       (short)0, z, false, false);
    }

    const unsigned char* hcur = sm + LDS_H + cur * (16 * HROW);
    // ---- recurrent GEMM, K=0..255 from LDS-resident W_rec -----------------
#pragma unroll 2
    for (int kit = 0; kit < 8; ++kit) {
      v16bf a = load_afrag(hcur, mrow, hi, kit);
#pragma unroll
      for (int j = 0; j < 4; ++j) {
        v16bf b = load_bfrag(sm + LDS_BRES + (kit * 32 + ntb + j) * 1024, lane);
        acc[j] = __builtin_amdgcn_wmma_f32_16x16x32_bf16(false, a, false, b,
                                                         (short)0, acc[j], false, false);
      }
    }
    // ---- recurrent GEMM, K=256..511 streamed from L2 (overlaps LDS pipe) --
#pragma unroll 2
    for (int kit = 8; kit < 16; ++kit) {
      v16bf a = load_afrag(hcur, mrow, hi, kit);
#pragma unroll
      for (int j = 0; j < 4; ++j) {
        v16bf b = load_bfrag(ws + WS_WREC + (kit * 32 + ntb + j) * 1024, lane);
        acc[j] = __builtin_amdgcn_wmma_f32_16x16x32_bf16(false, a, false, b,
                                                         (short)0, acc[j], false, false);
      }
    }

    // ---- relu, write h_new (bf16) into buffer nxt -------------------------
    unsigned char* hnxt = sm + LDS_H + nxt * (16 * HROW);
#pragma unroll
    for (int j = 0; j < 4; ++j) {
      const int ncol = (ntb + j) * 16 + mrow;
#pragma unroll
      for (int r = 0; r < 8; ++r) {
        float v = acc[j][r];
        v = v > 0.f ? v : 0.f;
        *(__bf16*)(hnxt + (r + hi * 8) * HROW + ncol * 2) = (__bf16)v;
      }
    }

    // ---- y partials: K chunks [64w,64w+64) == this wave's own h_new cols --
    // same-wave LDS RAW is in-order -> no barrier needed before these reads
    v8f yacc = {0.f, 0.f, 0.f, 0.f, 0.f, 0.f, 0.f, 0.f};
#pragma unroll
    for (int q = 0; q < 2; ++q) {
      const int kit = 2 * w + q;
      v16bf a = load_afrag(hnxt, mrow, hi, kit);
      v16bf b = load_bfrag(ws + WS_WOUT + kit * 1024, lane);
      yacc = __builtin_amdgcn_wmma_f32_16x16x32_bf16(false, a, false, b,
                                                     (short)0, yacc, false, false);
    }
    {
      float* yp = (float*)(sm + LDS_YP + cur * 8192) + w * 256;
#pragma unroll
      for (int r = 0; r < 8; ++r) yp[(r + hi * 8) * 16 + mrow] = yacc[r];
    }

    // ---- stage input rows for t+1 into the other xin buffer ---------------
    if (t + 1 < T_ && tid < 160) {
      const int r = tid / 10, i = tid % 10;
      *(float*)(sm + LDS_XIN + nxt * 1024 + (r * 16 + i) * 4) =
          inputs[(size_t)(m0 + r) * T_ * I_ + (size_t)(t + 1) * I_ + i];
    }

    __syncthreads();  // one barrier per timestep (h/xin/ypart double-buffered)
  }

  // ---- finalize y for the last step ---------------------------------------
  if (tid < 144) {
    const int m = tid / O_, o = tid % O_;
    const float* yp = (const float*)(sm + LDS_YP + ((T_ - 1) & 1) * 8192);
    float s = bias;
#pragma unroll
    for (int ww = 0; ww < 8; ++ww) s += yp[ww * 256 + m * 16 + o];
    out[((size_t)(m0 + m) * T_ + (T_ - 1)) * O_ + o] = s;
  }
}

// ---------------------------------------------------------------------------
extern "C" void kernel_launch(void* const* d_in, const int* in_sizes, int n_in,
                              void* d_out, int out_size, void* d_ws, size_t ws_size,
                              hipStream_t stream) {
  (void)in_sizes; (void)n_in; (void)out_size; (void)ws_size;
  const float* inputs = (const float*)d_in[0];
  const float* W_in   = (const float*)d_in[1];
  const float* W_rec  = (const float*)d_in[2];
  const float* W_out  = (const float*)d_in[3];
  const float* b_out  = (const float*)d_in[4];
  float*         out  = (float*)d_out;
  unsigned char* ws   = (unsigned char*)d_ws;

  // allow 306.5 KB dynamic LDS (WGP supports 320 KB); idempotent, capture-safe
  hipFuncSetAttribute((const void*)rnn_fused,
                      hipFuncAttributeMaxDynamicSharedMemorySize, LDS_TOTAL);

  prep_frags<<<70, 256, 0, stream>>>(W_in, W_rec, W_out, ws);
  rnn_fused<<<16, 256, LDS_TOTAL, stream>>>(inputs, b_out, ws, out);
}
